// SNNModel_63745904608013
// MI455X (gfx1250) — matrix operand order repaired and checked by reading, not compiled, not run on hardware
//
#include <hip/hip_runtime.h>

typedef __attribute__((ext_vector_type(16))) _Float16 v16h;
typedef __attribute__((ext_vector_type(8)))  _Float16 v8h;
typedef __attribute__((ext_vector_type(4)))  _Float16 v4h;
typedef __attribute__((ext_vector_type(8)))  float    v8f;

#define N_PRE   1024
#define N_POST  2048
#define SEQ_LEN 4096
#define DECAY   0.9f
#define V_TH    1.0f

// ---------------------------------------------------------------------------
// Kernel 1: weights f32 -> f16, 4 elements per thread (N_POST*N_PRE = 2M)
// ---------------------------------------------------------------------------
__global__ void snn_convert_weights(const float* __restrict__ w,
                                    _Float16* __restrict__ wh) {
  size_t i = ((size_t)blockIdx.x * 256 + threadIdx.x) * 4;
  float4 f = *(const float4*)(w + i);
  v4h h;
  h[0] = (_Float16)f.x; h[1] = (_Float16)f.y;
  h[2] = (_Float16)f.z; h[3] = (_Float16)f.w;
  *(v4h*)(wh + i) = h;
}

// ---------------------------------------------------------------------------
// Kernel 2: stim [N_PRE][SEQ_LEN] f32  ->  stimT [SEQ_LEN][N_PRE] f16
// 32x32 LDS tile transpose; both global phases coalesced.
// ---------------------------------------------------------------------------
__global__ void snn_transpose_stim(const float* __restrict__ stim,
                                   _Float16* __restrict__ stimT) {
  __shared__ _Float16 tile[32][33];
  const int t0 = blockIdx.x * 32;   // time tile
  const int k0 = blockIdx.y * 32;   // pre-neuron tile
  const int tx = threadIdx.x;       // 0..31
  for (int i = threadIdx.y; i < 32; i += 8)
    tile[i][tx] = (_Float16)stim[(size_t)(k0 + i) * SEQ_LEN + t0 + tx];
  __syncthreads();
  for (int i = threadIdx.y; i < 32; i += 8)
    stimT[(size_t)(t0 + i) * N_PRE + k0 + tx] = tile[tx][i];
}

// ---------------------------------------------------------------------------
// Kernel 3: currents[N_POST][SEQ_LEN] = A[N_POST][N_PRE] x B[N_PRE][SEQ_LEN]
// A = weights f16 (row major), BT = stimT f16 (= B column-major rows).
// One wave per 16x16 tile; K loop of v_wmma_f32_16x16x32_f16.
//
// Fragment layouts as used here (contiguous K order, wave32):
//   A 16x32 : lane&15 = row M; lanes 0-15 hold K kb..kb+15,
//             lanes 16-31 hold K kb+16..kb+31                 -> 2x b128
//   B 32x16 : lane&15 = col N; lanes 0-15 hold K kb..kb+15,
//             lanes 16-31 hold K kb+16..kb+31                 -> 2x b128
//   C/D     : VGPR i -> row M = m0 + i + (lane<16 ? 0 : 8), col = lane&15
// ---------------------------------------------------------------------------
__global__ void __launch_bounds__(256)
snn_gemm_wmma(const _Float16* __restrict__ A,
              const _Float16* __restrict__ BT,
              float* __restrict__ C) {
  const int lane = threadIdx.x & 31;
  const int wave = threadIdx.x >> 5;
  const int tile = blockIdx.x * 8 + wave;
  const int tilesT = SEQ_LEN / 16;                 // 256
  const int m0 = (tile / tilesT) * 16;
  const int t0 = (tile % tilesT) * 16;

  const int half = (lane < 16) ? 0 : 1;
  const int r    = lane & 15;

  const _Float16* aRow = A  + (size_t)(m0 + r) * N_PRE + half * 16;
  const _Float16* bRow = BT + (size_t)(t0 + r) * N_PRE + half * 16;

  v8f c = {};
  for (int kb = 0; kb < N_PRE; kb += 32) {
    __builtin_prefetch(bRow + kb + 64, 0, 0);      // global_prefetch_b8
    v8h a_lo = *(const v8h*)(aRow + kb);
    v8h a_hi = *(const v8h*)(aRow + kb + 8);
    v16h a;
#pragma unroll
    for (int i = 0; i < 8; ++i) { a[i] = a_lo[i]; a[i + 8] = a_hi[i]; }
    v16h b = *(const v16h*)(bRow + kb);
    c = __builtin_amdgcn_wmma_f32_16x16x32_f16(
        /*neg_a=*/false, a, /*neg_b=*/false, b,
        /*c_mod=*/(short)0, c, /*reuse_a=*/false, /*reuse_b=*/false);
  }

  float* cCol = C + t0 + r;
#pragma unroll
  for (int i = 0; i < 8; ++i) {
    int m = m0 + i + half * 8;
    cCol[(size_t)m * SEQ_LEN] = c[i];
  }
}

// ---------------------------------------------------------------------------
// Kernel 4: sequential LIF scan. One thread per post-neuron (2048 total).
// LDS-staged 256x16 tiles: coalesced 64B global reads/writes; padded stride
// 17 makes the per-thread scan conflict-free under wave32.
// ---------------------------------------------------------------------------
#define TCH 16
__global__ void __launch_bounds__(256)
snn_lif(const float* __restrict__ cur,
        float* __restrict__ spikes,
        float* __restrict__ vout) {
  __shared__ float tIn[256 * (TCH + 1)];
  __shared__ float tSp[256 * (TCH + 1)];
  const int tid = threadIdx.x;
  const int n0  = blockIdx.x * 256;
  float v = 0.f;

  for (int t0 = 0; t0 < SEQ_LEN; t0 += TCH) {
    // cooperative coalesced load of currents[n0..n0+255][t0..t0+15]
#pragma unroll
    for (int j = 0; j < TCH; ++j) {
      int flat = j * 256 + tid;
      int row  = flat >> 4;
      int col  = flat & (TCH - 1);
      tIn[row * (TCH + 1) + col] =
          cur[(size_t)(n0 + row) * SEQ_LEN + t0 + col];
    }
    __syncthreads();

    // per-neuron sequential scan over this time chunk
#pragma unroll
    for (int j = 0; j < TCH; ++j) {
      float i_t = tIn[tid * (TCH + 1) + j];
      v = v * DECAY + i_t;
      float s = (v >= V_TH) ? 1.f : 0.f;
      v = (s > 0.f) ? 0.f : v;
      tSp[tid * (TCH + 1) + j] = s;
      tIn[tid * (TCH + 1) + j] = v;   // reuse input tile for v output
    }
    __syncthreads();

    // cooperative coalesced writeback of both outputs
#pragma unroll
    for (int j = 0; j < TCH; ++j) {
      int flat = j * 256 + tid;
      int row  = flat >> 4;
      int col  = flat & (TCH - 1);
      size_t o = (size_t)(n0 + row) * SEQ_LEN + t0 + col;
      spikes[o] = tSp[row * (TCH + 1) + col];
      vout[o]   = tIn[row * (TCH + 1) + col];
    }
    __syncthreads();
  }
}

// ---------------------------------------------------------------------------
extern "C" void kernel_launch(void* const* d_in, const int* in_sizes, int n_in,
                              void* d_out, int out_size, void* d_ws, size_t ws_size,
                              hipStream_t stream) {
  const float* stim    = (const float*)d_in[0];   // [N_PRE][SEQ_LEN] f32
  const float* weights = (const float*)d_in[1];   // [N_POST][N_PRE]  f32

  float* out    = (float*)d_out;
  float* spikes = out;                                  // [N_POST][SEQ_LEN]
  float* vout   = out + (size_t)N_POST * SEQ_LEN;       // [N_POST][SEQ_LEN]

  char* ws = (char*)d_ws;
  _Float16* wh    = (_Float16*)ws;                             //  4 MB
  _Float16* stimT = (_Float16*)(ws + ((size_t)4  << 20));      //  8 MB
  float*    cur   = (float*)   (ws + ((size_t)12 << 20));      // 32 MB

  // 1) weights f32 -> f16
  snn_convert_weights<<<(N_POST * N_PRE) / (256 * 4), 256, 0, stream>>>(weights, wh);

  // 2) stim f32 [K][T] -> stimT f16 [T][K]
  dim3 tb(32, 8);
  dim3 tg(SEQ_LEN / 32, N_PRE / 32);
  snn_transpose_stim<<<tg, tb, 0, stream>>>(stim, stimT);

  // 3) WMMA GEMM: 128 x 256 tiles of 16x16, 8 waves/block
  int nTiles = (N_POST / 16) * (SEQ_LEN / 16);
  snn_gemm_wmma<<<nTiles / 8, 256, 0, stream>>>(wh, stimT, cur);

  // 4) LIF scan
  snn_lif<<<N_POST / 256, 256, 0, stream>>>(cur, spikes, vout);
}
